// Net_50319836839938
// MI455X (gfx1250) — compile-verified, matrix operand order checked by reading, not run on hardware
//
#include <hip/hip_runtime.h>
#include <hip/hip_bf16.h>
#include <math.h>
#include <stdint.h>

typedef __bf16 bf16;
typedef __attribute__((ext_vector_type(16))) __bf16 v16bf;
typedef __attribute__((ext_vector_type(8)))  float  v8f;

struct alignas(16) bf16x8 { bf16 v[8]; };

#define D_DIM  512
#define E_DIM  768
#define S_DIM  256
#define UVB_W  1792   // 2*E + S
#define NBUCK  64
#define NBLK   6

__device__ __forceinline__ float silu_f(float x){ return x / (1.f + __expf(-x)); }
__device__ __forceinline__ float gelu_f(float x){ return 0.5f * x * (1.f + erff(x * 0.7071067811865475f)); }

// CDNA5 async global->LDS copy (ASYNCcnt-tracked, GV addressing mode).
// Flat pointer low 32 bits are the LDS address (aperture rules, ISA 10.2).
__device__ __forceinline__ void async_copy_b128(void* lds_dst, const void* gsrc){
  unsigned lds = (unsigned)(uintptr_t)lds_dst;
  unsigned long long g = (unsigned long long)(uintptr_t)gsrc;
  asm volatile("global_load_async_to_lds_b128 %0, %1, off" :: "v"(lds), "v"(g) : "memory");
}
__device__ __forceinline__ void async_wait0(){
  asm volatile("s_wait_asynccnt 0x0" ::: "memory");
}

// ---------------------------------------------------------------- f32 -> bf16
__global__ void f2bf_kernel(const float* __restrict__ in, bf16* __restrict__ out, long n){
  long i = (long)blockIdx.x * blockDim.x + threadIdx.x;
  if (i < n) out[i] = (bf16)in[i];
}

// ---------------------------------------------------------------- WMMA GEMM
// C[M,N](f32) = epilogue( A[M,K](bf16,row) @ B[K,N](bf16,row) )
// mode 0: C = acc
// mode 1: C = silu(acc + bias[col])
// mode 2: C = Res + acc + bias[col]            (residual; Res may alias C)
// mode 3: C = Res + gelu(acc + bias[col])*(*alphaPtr)
// mode 4: C = acc + bias[col]
#define TM 128
#define TN 64
#define TK 32

__global__ __launch_bounds__(256)
void gemm_bf16_kernel(const bf16* __restrict__ A, const bf16* __restrict__ B,
                      float* __restrict__ C, const float* __restrict__ bias,
                      const float* __restrict__ Res, const float* __restrict__ alphaPtr,
                      int M, int N, int K, int mode)
{
  __shared__ bf16 As[TM][TK];   // 8 KB
  __shared__ bf16 Bs[TN][TK];   // 4 KB, K-transposed so fragments are ds_load_b128

  const int tid  = threadIdx.x;
  const int wave = tid >> 5, lane = tid & 31;
  const int half = lane >> 4, lm = lane & 15;
  const int kb0  = half * 8, kb1 = 16 + half * 8;
  const int rowBase = blockIdx.y * TM;
  const int colBase = blockIdx.x * TN;
  const bool fullA = (rowBase + TM <= M);   // uniform per block

  v8f zero = {0.f,0.f,0.f,0.f,0.f,0.f,0.f,0.f};
  v8f acc[4]; acc[0]=zero; acc[1]=zero; acc[2]=zero; acc[3]=zero;

  for (int k0 = 0; k0 < K; k0 += TK){
    // ---- stage A tile (128x32 bf16 = 512 x 16B chunks, 2 per thread) ----
    if (fullA){
      for (int ci = tid; ci < (TM*TK)/8; ci += 256){
        int r = ci >> 2, c8 = (ci & 3) * 8;
        async_copy_b128(&As[r][c8], A + (size_t)(rowBase + r) * K + k0 + c8);
      }
    } else {
      for (int ci = tid; ci < (TM*TK)/8; ci += 256){
        int r = ci >> 2, c8 = (ci & 3) * 8;
        int gr = rowBase + r;
        bf16x8 w;
        if (gr < M){
          w = *(const bf16x8*)(A + (size_t)gr * K + k0 + c8);
        } else {
          for (int i = 0; i < 8; i++) w.v[i] = (bf16)0.f;
        }
        *(bf16x8*)&As[r][c8] = w;
      }
    }
    // ---- stage B tile transposed (32x64 bf16 = 256 x 16B chunks, 1/thread) ----
    {
      int ci = tid;                    // (TN*TK)/8 == 256 == blockDim
      int kk = ci >> 3, n0 = (ci & 7) * 8;
      bf16x8 w = *(const bf16x8*)(B + (size_t)(k0 + kk) * N + colBase + n0);
      #pragma unroll
      for (int i = 0; i < 8; i++) Bs[n0 + i][kk] = w.v[i];
    }
    if (k0 + TK < K)                   // prefetch next B tile row
      __builtin_prefetch(B + (size_t)(k0 + TK + (tid & 31)) * N + colBase, 0, 1);
    if (fullA) async_wait0();
    __syncthreads();

    v16bf af;
    #pragma unroll
    for (int i = 0; i < 8; i++){
      af[i]   = As[wave*16 + lm][kb0 + i];
      af[8+i] = As[wave*16 + lm][kb1 + i];
    }
    #pragma unroll
    for (int t = 0; t < 4; t++){
      v16bf bfv;
      #pragma unroll
      for (int i = 0; i < 8; i++){
        bfv[i]   = Bs[t*16 + lm][kb0 + i];
        bfv[8+i] = Bs[t*16 + lm][kb1 + i];
      }
      acc[t] = __builtin_amdgcn_wmma_f32_16x16x32_bf16(false, af, false, bfv,
                                                       (short)0, acc[t], false, false);
    }
    __syncthreads();
  }

  float alpha = (mode == 3) ? alphaPtr[0] : 0.f;
  #pragma unroll
  for (int t = 0; t < 4; t++){
    int col = colBase + t*16 + lm;
    #pragma unroll
    for (int r = 0; r < 8; r++){
      int row = rowBase + wave*16 + half*8 + r;
      if (row < M){
        size_t o = (size_t)row * N + col;
        float v = acc[t][r];
        if      (mode == 0) C[o] = v;
        else if (mode == 1) C[o] = silu_f(v + bias[col]);
        else if (mode == 2) C[o] = Res[o] + v + bias[col];
        else if (mode == 3) C[o] = Res[o] + gelu_f(v + bias[col]) * alpha;
        else                C[o] = v + bias[col];
      }
    }
  }
}

// ---------------------------------------------------------------- attention
// One block per bucket of 64 tokens. uvb row = [u(768) | v(768) | base(256)].
// q = base*gam0+bet0, k = base*gam1+bet1; a = relu(q k^T / 64)^2; o = (a@v)*u
__global__ __launch_bounds__(512)
void attn_kernel(const float* __restrict__ uvb, const float* __restrict__ gam,
                 const float* __restrict__ bet, bf16* __restrict__ obf)
{
  __shared__ bf16 ks[64][S_DIM];   // 32 KB
  __shared__ bf16 as_[64][64];     //  8 KB
  const size_t rowbase = (size_t)blockIdx.x * 64;
  const int tid = threadIdx.x;
  const float* gam0 = gam;          const float* gam1 = gam + S_DIM;
  const float* bet0 = bet;          const float* bet1 = bet + S_DIM;

  for (int i = tid; i < 64*S_DIM; i += 512){
    int l = i >> 8, s = i & 255;
    float b = uvb[(rowbase + l)*UVB_W + 2*E_DIM + s];
    ks[l][s] = (bf16)(b * gam1[s] + bet1[s]);
  }
  __syncthreads();

  const int wave = tid >> 5, lane = tid & 31;
  const int half = lane >> 4, lm = lane & 15;
  const int kb0 = half*8, kb1 = 16 + half*8;

  { // a = relu(q k^T / 64)^2 : 4x4 grid of 16x16 tiles, one per wave
    int tr = wave & 3, tc = wave >> 2;
    v8f acc = {0.f,0.f,0.f,0.f,0.f,0.f,0.f,0.f};
    const float* qrow = uvb + (rowbase + tr*16 + lm)*UVB_W + 2*E_DIM;
    for (int k0 = 0; k0 < S_DIM; k0 += 32){
      v16bf af, bfv;
      #pragma unroll
      for (int i = 0; i < 8; i++){
        int s0 = k0 + kb0 + i, s1 = k0 + kb1 + i;
        af[i]    = (bf16)(qrow[s0]*gam0[s0] + bet0[s0]);
        af[8+i]  = (bf16)(qrow[s1]*gam0[s1] + bet0[s1]);
        bfv[i]   = ks[tc*16 + lm][s0];
        bfv[8+i] = ks[tc*16 + lm][s1];
      }
      acc = __builtin_amdgcn_wmma_f32_16x16x32_bf16(false, af, false, bfv,
                                                    (short)0, acc, false, false);
    }
    #pragma unroll
    for (int r = 0; r < 8; r++){
      float a = acc[r] * (1.f/64.f);
      a = a > 0.f ? a : 0.f;
      as_[tr*16 + half*8 + r][tc*16 + lm] = (bf16)(a*a);
    }
  }
  __syncthreads();

  { // o = a @ v (M=64,N=768,K=64), fused *u, bf16 out. 12 col tiles per wave.
    int rw = wave & 3, cg = wave >> 2;
    for (int cti = 0; cti < 12; cti++){
      int ct = cg*12 + cti;
      v8f acc = {0.f,0.f,0.f,0.f,0.f,0.f,0.f,0.f};
      for (int k0 = 0; k0 < 64; k0 += 32){
        v16bf af, bfv;
        #pragma unroll
        for (int i = 0; i < 8; i++){
          af[i]    = as_[rw*16 + lm][k0 + kb0 + i];
          af[8+i]  = as_[rw*16 + lm][k0 + kb1 + i];
          bfv[i]   = (bf16)uvb[(rowbase + k0 + kb0 + i)*UVB_W + E_DIM + ct*16 + lm];
          bfv[8+i] = (bf16)uvb[(rowbase + k0 + kb1 + i)*UVB_W + E_DIM + ct*16 + lm];
        }
        acc = __builtin_amdgcn_wmma_f32_16x16x32_bf16(false, af, false, bfv,
                                                      (short)0, acc, false, false);
      }
      #pragma unroll
      for (int r = 0; r < 8; r++){
        int row = rw*16 + half*8 + r, col = ct*16 + lm;
        float u = uvb[(rowbase + row)*UVB_W + col];
        obf[(rowbase + row)*E_DIM + col] = (bf16)(acc[r] * u);
      }
    }
  }
}

// ---------------------------------------------------------------- layernorm (wave per row, W=512)
__global__ __launch_bounds__(256)
void ln_leaky_kernel(float* __restrict__ x, const float* __restrict__ g,
                     const float* __restrict__ b, int M, float eps)
{
  int row = blockIdx.x * 8 + (threadIdx.x >> 5);
  int lane = threadIdx.x & 31;
  if (row >= M) return;
  float* xr = x + (size_t)row * D_DIM;
  float v[16]; float s = 0.f;
  #pragma unroll
  for (int i = 0; i < 16; i++){ v[i] = xr[lane + 32*i]; s += v[i]; }
  #pragma unroll
  for (int m = 16; m > 0; m >>= 1) s += __shfl_xor(s, m, 32);
  float mean = s * (1.f/512.f);
  float q = 0.f;
  #pragma unroll
  for (int i = 0; i < 16; i++){ float d = v[i]-mean; q += d*d; }
  #pragma unroll
  for (int m = 16; m > 0; m >>= 1) q += __shfl_xor(q, m, 32);
  float rstd = rsqrtf(q*(1.f/512.f) + eps);
  #pragma unroll
  for (int i = 0; i < 16; i++){
    int c = lane + 32*i;
    float h = (v[i]-mean)*rstd*g[c] + b[c];
    xr[c] = h > 0.f ? h : 0.1f*h;
  }
}

__global__ __launch_bounds__(256)
void ln_bf16_kernel(const float* __restrict__ x, const float* __restrict__ g,
                    const float* __restrict__ b, bf16* __restrict__ out, int M, float eps)
{
  int row = blockIdx.x * 8 + (threadIdx.x >> 5);
  int lane = threadIdx.x & 31;
  if (row >= M) return;
  const float* xr = x + (size_t)row * D_DIM;
  bf16* orow = out + (size_t)row * D_DIM;
  float v[16]; float s = 0.f;
  #pragma unroll
  for (int i = 0; i < 16; i++){ v[i] = xr[lane + 32*i]; s += v[i]; }
  #pragma unroll
  for (int m = 16; m > 0; m >>= 1) s += __shfl_xor(s, m, 32);
  float mean = s * (1.f/512.f);
  float q = 0.f;
  #pragma unroll
  for (int i = 0; i < 16; i++){ float d = v[i]-mean; q += d*d; }
  #pragma unroll
  for (int m = 16; m > 0; m >>= 1) q += __shfl_xor(q, m, 32);
  float rstd = rsqrtf(q*(1.f/512.f) + eps);
  #pragma unroll
  for (int i = 0; i < 16; i++){
    int c = lane + 32*i;
    orow[c] = (bf16)((v[i]-mean)*rstd*g[c] + b[c]);
  }
}

// ---------------------------------------------------------------- sort machinery
__global__ __launch_bounds__(256)
void norms_kernel(const float* __restrict__ x, float* __restrict__ out, int M){
  int row = blockIdx.x * 8 + (threadIdx.x >> 5);
  int lane = threadIdx.x & 31;
  if (row >= M) return;
  const float* xr = x + (size_t)row * D_DIM;
  float s = 0.f;
  for (int i = lane; i < D_DIM; i += 32){ float v = xr[i]; s += v*v; }
  for (int m = 16; m > 0; m >>= 1) s += __shfl_xor(s, m, 32);
  if (lane == 0) out[row] = s;   // sumsq: monotone in L2 norm
}

// bitonic argsort (ascending), one block per sample, L power-of-two <= 4096
__global__ __launch_bounds__(1024)
void sort_kernel(const float* __restrict__ norms, int* __restrict__ ids, int L){
  __shared__ float key[4096];
  __shared__ int   vid[4096];
  const float* nrm = norms + (size_t)blockIdx.x * L;
  int* out = ids + (size_t)blockIdx.x * L;
  for (int i = threadIdx.x; i < L; i += blockDim.x){ key[i] = nrm[i]; vid[i] = i; }
  __syncthreads();
  for (int k = 2; k <= L; k <<= 1){
    for (int j = k >> 1; j > 0; j >>= 1){
      for (int i = threadIdx.x; i < L; i += blockDim.x){
        int ixj = i ^ j;
        if (ixj > i){
          bool dir = ((i & k) == 0);
          float a = key[i], b = key[ixj];
          if ((a > b) == dir){
            key[i] = b; key[ixj] = a;
            int t = vid[i]; vid[i] = vid[ixj]; vid[ixj] = t;
          }
        }
      }
      __syncthreads();
    }
  }
  for (int i = threadIdx.x; i < L; i += blockDim.x) out[i] = vid[i];
}

__global__ void gather_kernel(const float* __restrict__ seq, const int* __restrict__ ids,
                              float* __restrict__ out, int L){
  int row = blockIdx.x;
  int sample = row / L, local = row % L;
  int src = sample * L + ids[sample * L + local];
  const float* s = seq + (size_t)src * D_DIM;
  float* o = out + (size_t)row * D_DIM;
  for (int d = threadIdx.x; d < D_DIM; d += blockDim.x) o[d] = s[d];
}

// ---------------------------------------------------------------- reductions
__global__ void bucket_mean_kernel(const float* __restrict__ hb, float* __restrict__ gy,
                                   bf16* __restrict__ gybf){
  int g = blockIdx.x, d = threadIdx.x;           // blockDim = 512
  float s = 0.f;
  for (int l = 0; l < NBUCK; l++) s += hb[((size_t)g*NBUCK + l)*D_DIM + d];
  s *= (1.f/64.f);
  gy[(size_t)g*D_DIM + d] = s;
  gybf[(size_t)g*D_DIM + d] = (bf16)s;
}

__global__ void sample_mean_kernel(const float* __restrict__ x, float* __restrict__ out, int R){
  int b = blockIdx.x, d = threadIdx.x;           // blockDim = 512
  float s = 0.f;
  for (int r = 0; r < R; r++) s += x[((size_t)b*R + r)*D_DIM + d];
  out[(size_t)b*D_DIM + d] = s / (float)R;
}

__global__ void combine_kernel(const float* __restrict__ a, const float* __restrict__ b,
                               float* __restrict__ y, bf16* __restrict__ ybf){
  int i = blockIdx.x * 512 + threadIdx.x;
  float v = 0.5f * (a[i] + b[i]);
  y[i] = v; ybf[i] = (bf16)v;
}

__global__ __launch_bounds__(32)
void out_kernel(const float* __restrict__ y, const float* __restrict__ W,
                const float* __restrict__ bvec, float* __restrict__ out){
  int b = blockIdx.x >> 1, j = blockIdx.x & 1;
  float s = 0.f;
  for (int d = threadIdx.x; d < D_DIM; d += 32) s += y[(size_t)b*D_DIM + d] * W[d*2 + j];
  for (int m = 16; m > 0; m >>= 1) s += __shfl_xor(s, m, 32);
  if (threadIdx.x == 0) out[b*2 + j] = s + bvec[j];
}

// ---------------------------------------------------------------- driver
extern "C" void kernel_launch(void* const* d_in, const int* in_sizes, int n_in,
                              void* d_out, int out_size, void* d_ws, size_t ws_size,
                              hipStream_t stream)
{
  (void)in_sizes; (void)n_in; (void)out_size; (void)ws_size;
  const float* xs      = (const float*)d_in[0];
  const float* W_in    = (const float*)d_in[1];
  const float* b_in    = (const float*)d_in[2];
  const float* ln_in_g = (const float*)d_in[3];
  const float* ln_in_b = (const float*)d_in[4];
  const float* blk_ln_g= (const float*)d_in[5];
  const float* blk_ln_b= (const float*)d_in[6];
  const float* blk_Wuv = (const float*)d_in[7];
  const float* blk_buv = (const float*)d_in[8];
  const float* blk_gam = (const float*)d_in[9];
  const float* blk_bet = (const float*)d_in[10];
  const float* blk_Wo  = (const float*)d_in[11];
  const float* blk_bo  = (const float*)d_in[12];
  const float* tr_W    = (const float*)d_in[13];
  const float* tr_b    = (const float*)d_in[14];
  const float* tr_a    = (const float*)d_in[15];
  const float* tr2_W   = (const float*)d_in[16];
  const float* tr2_b   = (const float*)d_in[17];
  const float* tr2_a   = (const float*)d_in[18];
  const float* out_W   = (const float*)d_in[19];
  const float* out_b   = (const float*)d_in[20];
  float* outp = (float*)d_out;

  char* base = (char*)d_ws; size_t off = 0;
  auto alloc = [&](size_t bytes)->void*{
    void* p = base + off; off = (off + bytes + 255) & ~(size_t)255; return p;
  };
  auto cdiv = [](long a, long b)->int{ return (int)((a + b - 1) / b); };

  const int B = 4, N0 = 4096, IN = 768;
  const long nX = (long)B * N0 * IN;
  const int M1 = B * N0;                          // 16384 rows

  bf16* Xbf    = (bf16*)alloc((size_t)nX * 2);
  bf16* Winbf  = (bf16*)alloc((size_t)IN * D_DIM * 2);
  bf16* Wuvbf  = (bf16*)alloc((size_t)NBLK * D_DIM * UVB_W * 2);
  bf16* Wobf   = (bf16*)alloc((size_t)NBLK * E_DIM * D_DIM * 2);
  bf16* trWbf  = (bf16*)alloc((size_t)D_DIM * D_DIM * 2);
  bf16* tr2Wbf = (bf16*)alloc((size_t)D_DIM * D_DIM * 2);
  float* h0    = (float*)alloc((size_t)M1 * D_DIM * 4);
  float* hb    = (float*)alloc((size_t)M1 * D_DIM * 4);
  bf16*  nbf   = (bf16*) alloc((size_t)M1 * D_DIM * 2);
  float* uvb   = (float*)alloc((size_t)M1 * UVB_W * 4);
  bf16*  obf   = (bf16*) alloc((size_t)M1 * E_DIM * 2);
  float* norms = (float*)alloc((size_t)M1 * 4);
  int*   ids   = (int*)  alloc((size_t)M1 * 4);
  float* gy    = (float*)alloc((size_t)B * 64 * D_DIM * 4);
  bf16*  gybf  = (bf16*) alloc((size_t)B * 64 * D_DIM * 2);
  float* trout = (float*)alloc((size_t)B * 64 * D_DIM * 4);
  float* ys0   = (float*)alloc((size_t)B * D_DIM * 4);
  float* hb2   = (float*)alloc((size_t)B * 64 * D_DIM * 4);
  float* gy2   = (float*)alloc((size_t)B * D_DIM * 4);
  bf16*  gy2bf = (bf16*) alloc((size_t)B * D_DIM * 2);
  float* trout2= (float*)alloc((size_t)B * D_DIM * 4);
  float* ys1   = (float*)alloc((size_t)B * D_DIM * 4);
  float* yc    = (float*)alloc((size_t)B * D_DIM * 4);
  bf16*  ybf   = (bf16*) alloc((size_t)B * D_DIM * 2);
  float* y2    = (float*)alloc((size_t)B * D_DIM * 4);

  // bf16 conversions (every call: deterministic, no caching)
  f2bf_kernel<<<cdiv(nX,256),256,0,stream>>>(xs, Xbf, nX);
  f2bf_kernel<<<cdiv((long)IN*D_DIM,256),256,0,stream>>>(W_in, Winbf, (long)IN*D_DIM);
  f2bf_kernel<<<cdiv((long)NBLK*D_DIM*UVB_W,256),256,0,stream>>>(blk_Wuv, Wuvbf, (long)NBLK*D_DIM*UVB_W);
  f2bf_kernel<<<cdiv((long)NBLK*E_DIM*D_DIM,256),256,0,stream>>>(blk_Wo, Wobf, (long)NBLK*E_DIM*D_DIM);
  f2bf_kernel<<<cdiv((long)D_DIM*D_DIM,256),256,0,stream>>>(tr_W,  trWbf,  (long)D_DIM*D_DIM);
  f2bf_kernel<<<cdiv((long)D_DIM*D_DIM,256),256,0,stream>>>(tr2_W, tr2Wbf, (long)D_DIM*D_DIM);

  // h0 = leaky_relu(layernorm(X @ W_in + b_in))
  gemm_bf16_kernel<<<dim3(D_DIM/TN, cdiv(M1,TM)),256,0,stream>>>(
      Xbf, Winbf, h0, b_in, nullptr, nullptr, M1, D_DIM, IN, 4);
  ln_leaky_kernel<<<cdiv(M1,8),256,0,stream>>>(h0, ln_in_g, ln_in_b, M1, 1e-8f);

  // ---------------- iteration 1: L=4096 per sample, 256 buckets ----------------
  {
    int L = N0, Mr = B*L, nb = Mr/NBUCK;
    norms_kernel<<<cdiv(Mr,8),256,0,stream>>>(h0, norms, Mr);
    sort_kernel<<<B,1024,0,stream>>>(norms, ids, L);
    gather_kernel<<<Mr,128,0,stream>>>(h0, ids, hb, L);
    for (int blk = 0; blk < NBLK; blk++){
      ln_bf16_kernel<<<cdiv(Mr,8),256,0,stream>>>(
          hb, blk_ln_g + blk*D_DIM, blk_ln_b + blk*D_DIM, nbf, Mr, 1e-5f);
      gemm_bf16_kernel<<<dim3(UVB_W/TN, cdiv(Mr,TM)),256,0,stream>>>(
          nbf, Wuvbf + (size_t)blk*D_DIM*UVB_W, uvb,
          blk_buv + blk*UVB_W, nullptr, nullptr, Mr, UVB_W, D_DIM, 1);
      attn_kernel<<<nb,512,0,stream>>>(
          uvb, blk_gam + blk*2*S_DIM, blk_bet + blk*2*S_DIM, obf);
      gemm_bf16_kernel<<<dim3(D_DIM/TN, cdiv(Mr,TM)),256,0,stream>>>(
          obf, Wobf + (size_t)blk*E_DIM*D_DIM, hb,
          blk_bo + blk*D_DIM, hb, nullptr, Mr, D_DIM, E_DIM, 2);
    }
    bucket_mean_kernel<<<nb,512,0,stream>>>(hb, gy, gybf);
    gemm_bf16_kernel<<<dim3(D_DIM/TN, cdiv(nb,TM)),256,0,stream>>>(
        gybf, trWbf, trout, tr_b, gy, tr_a, nb, D_DIM, D_DIM, 3);
    sample_mean_kernel<<<B,512,0,stream>>>(trout, ys0, 64);
  }
  // ---------------- iteration 2: L=64 per sample, 4 buckets ----------------
  {
    int L = 64, Mr = B*L, nb = Mr/NBUCK;
    norms_kernel<<<cdiv(Mr,8),256,0,stream>>>(gy, norms, Mr);
    sort_kernel<<<B,1024,0,stream>>>(norms, ids, L);
    gather_kernel<<<Mr,128,0,stream>>>(gy, ids, hb2, L);
    for (int blk = 0; blk < NBLK; blk++){
      ln_bf16_kernel<<<cdiv(Mr,8),256,0,stream>>>(
          hb2, blk_ln_g + blk*D_DIM, blk_ln_b + blk*D_DIM, nbf, Mr, 1e-5f);
      gemm_bf16_kernel<<<dim3(UVB_W/TN, cdiv(Mr,TM)),256,0,stream>>>(
          nbf, Wuvbf + (size_t)blk*D_DIM*UVB_W, uvb,
          blk_buv + blk*UVB_W, nullptr, nullptr, Mr, UVB_W, D_DIM, 1);
      attn_kernel<<<nb,512,0,stream>>>(
          uvb, blk_gam + blk*2*S_DIM, blk_bet + blk*2*S_DIM, obf);
      gemm_bf16_kernel<<<dim3(D_DIM/TN, cdiv(Mr,TM)),256,0,stream>>>(
          obf, Wobf + (size_t)blk*E_DIM*D_DIM, hb2,
          blk_bo + blk*D_DIM, hb2, nullptr, Mr, D_DIM, E_DIM, 2);
    }
    bucket_mean_kernel<<<nb,512,0,stream>>>(hb2, gy2, gy2bf);
    gemm_bf16_kernel<<<dim3(D_DIM/TN,1),256,0,stream>>>(
        gy2bf, trWbf, trout2, tr_b, gy2, tr_a, nb, D_DIM, D_DIM, 3);
    sample_mean_kernel<<<B,512,0,stream>>>(trout2, ys1, 1);
  }
  // y = tr_layer2(mean(ys)); out = y @ out_W + out_b
  combine_kernel<<<B,512,0,stream>>>(ys0, ys1, yc, ybf);
  gemm_bf16_kernel<<<dim3(D_DIM/TN,1),256,0,stream>>>(
      ybf, tr2Wbf, y2, tr2_b, yc, tr2_a, B, D_DIM, D_DIM, 3);
  out_kernel<<<B*2,32,0,stream>>>(y2, out_W, out_b, outp);
}
